// EncoderLayer_30090540876475
// MI455X (gfx1250) — matrix operand order repaired and checked by reading, not compiled, not run on hardware
//
#include <hip/hip_runtime.h>
#include <math.h>

// ---------------------------------------------------------------------------
// CDNA5 / gfx1250 transformer encoder layer, bf16x3 split precision everywhere
// (fp32-class accuracy), all matrix math on V_WMMA_F32_16X16X32_BF16.
//
// bf16 16x16x32 fragment layouts (CDNA5 ISA 7.12.2), hi = lane>>4:
//   A (16x32): lane M = lane&15; contiguous K runs [8*hi,8*hi+8) and
//              [16+8*hi, 16+8*hi+8)            -> two 16B reads
//   B (32x16): lane N = lane&15; contiguous K runs [8*hi,8*hi+8) and
//              [16+8*hi, 16+8*hi+8)            -> two 16B reads
//   C/D 16x16: VGPR g -> row g (lanes 0-15) / g+8 (lanes 16-31), col = lane&15
//
// Split: x = hi + lo, hi = truncate-to-bf16(x) (residual exact in f32),
// lo = rne-bf16(x - hi).  x*y ~= hi*hi' + hi*lo' + lo*hi'  (err ~2^-16 rel).
// ---------------------------------------------------------------------------

typedef __attribute__((ext_vector_type(8)))  float  v8f;
typedef __attribute__((ext_vector_type(16))) __bf16 v16bf;
typedef __attribute__((ext_vector_type(4)))  unsigned int u32x4;

#define WMMA_BF16(A, B, C) \
    __builtin_amdgcn_wmma_f32_16x16x32_bf16(false, (A), false, (B), (short)0, (C), false, false)

static constexpr int BDIM = 1024;      // D_MODEL
static constexpr int SEQ  = 2048;
static constexpr int NB   = 2;
static constexpr int NH   = 16;
static constexpr int DK   = 64;
static constexpr int DFF  = 4096;
static constexpr int MROWS = NB * SEQ; // 4096 tokens

union Frag16 {
    v16bf v;
    u32x4 q[2];
};

__device__ __forceinline__ unsigned bf16_rne_bits(float x) {
    unsigned u = __float_as_uint(x);
    unsigned r = u + 0x7FFFu + ((u >> 16) & 1u);
    return r >> 16;
}

// gelu_tanh(x) = x * sigmoid(2c(x + 0.044715 x^3)) -- one hw exp + one divide.
__device__ __forceinline__ float gelu_tanh_f(float x) {
    const float c2 = 1.5957691216057308f; // 2*sqrt(2/pi)
    float u = c2 * (x + 0.044715f * x * x * x);
    u = fminf(fmaxf(u, -80.0f), 80.0f);
    float t = __expf(u);
    return x * t / (t + 1.0f);
}

// ---------------------------------------------------------------------------
// LayerNorm: unbiased std (n-1), eps added to std (matches reference).
// ---------------------------------------------------------------------------
__global__ __launch_bounds__(256) void ln_kernel(const float* __restrict__ x,
                                                 const float* __restrict__ gamma,
                                                 const float* __restrict__ beta,
                                                 float* __restrict__ out) {
    __shared__ float sm[8];
    const int row = blockIdx.x;
    const int t = threadIdx.x;
    const float* xr = x + (size_t)row * BDIM;

    float4 v = *(const float4*)(xr + t * 4);
    float s = v.x + v.y + v.z + v.w;
#pragma unroll
    for (int m = 16; m > 0; m >>= 1) s += __shfl_xor(s, m, 32);
    if ((t & 31) == 0) sm[t >> 5] = s;
    __syncthreads();
    if (t == 0) {
        float a = 0.f;
#pragma unroll
        for (int i = 0; i < 8; ++i) a += sm[i];
        sm[0] = a;
    }
    __syncthreads();
    const float mean = sm[0] * (1.0f / (float)BDIM);
    __syncthreads();

    const float dx = v.x - mean, dy = v.y - mean, dz = v.z - mean, dw = v.w - mean;
    float s2 = dx * dx + dy * dy + dz * dz + dw * dw;
#pragma unroll
    for (int m = 16; m > 0; m >>= 1) s2 += __shfl_xor(s2, m, 32);
    if ((t & 31) == 0) sm[t >> 5] = s2;
    __syncthreads();
    if (t == 0) {
        float a = 0.f;
#pragma unroll
        for (int i = 0; i < 8; ++i) a += sm[i];
        sm[0] = a;
    }
    __syncthreads();
    const float var = sm[0] * (1.0f / (float)(BDIM - 1));
    const float inv = 1.0f / (sqrtf(var) + 1e-6f);

    float4 g4 = *(const float4*)(gamma + t * 4);
    float4 b4 = *(const float4*)(beta + t * 4);
    float4 o;
    o.x = g4.x * dx * inv + b4.x;
    o.y = g4.y * dy * inv + b4.y;
    o.z = g4.z * dz * inv + b4.z;
    o.w = g4.w * dw * inv + b4.w;
    *(float4*)(out + (size_t)row * BDIM + t * 4) = o;
}

// ---------------------------------------------------------------------------
// bf16x3 GEMM: act(A@W + bias (+res)) with selectable output form.
//   OMODE 0: fp32 C            OMODE 1: split bf16 (CH,CL) row-major
//   OMODE 2: split bf16 transposed per head: [b][head][dim][seq]  (for V)
// Block tile 128x64, 8 waves x (32x32 slab), BK=32, double-buffered LDS.
// ---------------------------------------------------------------------------
template <int ACT, bool RES, int OMODE>
__global__ __launch_bounds__(256) void gemm_bf16x3_kernel(const float* __restrict__ A,
                                                          const float* __restrict__ W,
                                                          const float* __restrict__ bias,
                                                          const float* __restrict__ res,
                                                          float* __restrict__ C,
                                                          unsigned short* __restrict__ CH,
                                                          unsigned short* __restrict__ CL,
                                                          int M, int N, int K) {
    // rows padded to 40 bf16 (80 B): 16B-aligned uint4 frag reads, conflict-free.
    __shared__ __align__(16) unsigned short Ah[2][128][40];
    __shared__ __align__(16) unsigned short Al[2][128][40];
    __shared__ __align__(16) unsigned short Bh[2][64][40];  // transposed: [n][k]
    __shared__ __align__(16) unsigned short Bl[2][64][40];

    const int tid = threadIdx.x;
    const int lane = tid & 31, wave = tid >> 5;
    const int lo = lane & 15, hi = lane >> 4;
    const int m0 = (wave & 3) * 32;
    const int n0 = (wave >> 2) * 32;
    const int bm = blockIdx.y * 128;
    const int bn = blockIdx.x * 64;

    const int am = tid >> 1;
    const int akc = (tid & 1) * 16;
    const int bcol = tid & 63;
    const int bkq = (tid >> 6) * 8;

    float4 areg[4];
    float breg[8];

    auto load_slab = [&](int s) {
        const float* ap = A + (size_t)(bm + am) * K + s * 32 + akc;
#pragma unroll
        for (int i = 0; i < 4; ++i) areg[i] = *(const float4*)(ap + i * 4);
        const float* wp = W + (size_t)(s * 32 + bkq) * N + bn + bcol;
#pragma unroll
        for (int i = 0; i < 8; ++i) breg[i] = wp[(size_t)i * N];
    };

    auto store_slab = [&](int buf) {
#pragma unroll
        for (int i = 0; i < 4; ++i) {
            const float4 a = areg[i];
            unsigned hx = __float_as_uint(a.x) >> 16;
            unsigned hy = __float_as_uint(a.y) >> 16;
            unsigned hz = __float_as_uint(a.z) >> 16;
            unsigned hw = __float_as_uint(a.w) >> 16;
            float lx = a.x - __uint_as_float(hx << 16);
            float ly = a.y - __uint_as_float(hy << 16);
            float lz = a.z - __uint_as_float(hz << 16);
            float lw = a.w - __uint_as_float(hw << 16);
            uint2 ph, pl;
            ph.x = hx | (hy << 16);
            ph.y = hz | (hw << 16);
            pl.x = bf16_rne_bits(lx) | (bf16_rne_bits(ly) << 16);
            pl.y = bf16_rne_bits(lz) | (bf16_rne_bits(lw) << 16);
            *(uint2*)(&Ah[buf][am][akc + i * 4]) = ph;
            *(uint2*)(&Al[buf][am][akc + i * 4]) = pl;
        }
        unsigned h[8], l[8];
#pragma unroll
        for (int i = 0; i < 8; ++i) {
            unsigned u = __float_as_uint(breg[i]);
            h[i] = u >> 16;
            l[i] = bf16_rne_bits(breg[i] - __uint_as_float(u & 0xFFFF0000u));
        }
        uint4 qh = make_uint4(h[0] | (h[1] << 16), h[2] | (h[3] << 16),
                              h[4] | (h[5] << 16), h[6] | (h[7] << 16));
        uint4 ql = make_uint4(l[0] | (l[1] << 16), l[2] | (l[3] << 16),
                              l[4] | (l[5] << 16), l[6] | (l[7] << 16));
        *(uint4*)(&Bh[buf][bcol][bkq]) = qh;
        *(uint4*)(&Bl[buf][bcol][bkq]) = ql;
    };

    v8f acc[2][2] = {{{}, {}}, {{}, {}}};

    const int NS = K >> 5;
    load_slab(0);
    store_slab(0);

    for (int s = 0; s < NS; ++s) {
        __syncthreads();
        const int cur = s & 1;
        if (s + 1 < NS) load_slab(s + 1);

        Frag16 ah[2], al[2], bh[2], bl[2];
#pragma unroll
        for (int mi = 0; mi < 2; ++mi) {
            const unsigned short* ph = &Ah[cur][m0 + mi * 16 + lo][8 * hi];
            const unsigned short* pl = &Al[cur][m0 + mi * 16 + lo][8 * hi];
            ah[mi].q[0] = *(const u32x4*)ph;
            ah[mi].q[1] = *(const u32x4*)(ph + 16);
            al[mi].q[0] = *(const u32x4*)pl;
            al[mi].q[1] = *(const u32x4*)(pl + 16);
        }
#pragma unroll
        for (int ni = 0; ni < 2; ++ni) {
            const unsigned short* ph = &Bh[cur][n0 + ni * 16 + lo][8 * hi];
            const unsigned short* pl = &Bl[cur][n0 + ni * 16 + lo][8 * hi];
            bh[ni].q[0] = *(const u32x4*)ph;
            bh[ni].q[1] = *(const u32x4*)(ph + 16);
            bl[ni].q[0] = *(const u32x4*)pl;
            bl[ni].q[1] = *(const u32x4*)(pl + 16);
        }
#pragma unroll
        for (int mi = 0; mi < 2; ++mi)
#pragma unroll
            for (int ni = 0; ni < 2; ++ni)
                acc[mi][ni] = WMMA_BF16(ah[mi].v, bh[ni].v, acc[mi][ni]);
#pragma unroll
        for (int mi = 0; mi < 2; ++mi)
#pragma unroll
            for (int ni = 0; ni < 2; ++ni)
                acc[mi][ni] = WMMA_BF16(al[mi].v, bh[ni].v, acc[mi][ni]);
#pragma unroll
        for (int mi = 0; mi < 2; ++mi)
#pragma unroll
            for (int ni = 0; ni < 2; ++ni)
                acc[mi][ni] = WMMA_BF16(ah[mi].v, bl[ni].v, acc[mi][ni]);

        if (s + 1 < NS) store_slab(cur ^ 1);
    }

    const int c0 = bn + n0 + lo;
    const float bia0 = bias[c0];
    const float bia1 = bias[c0 + 16];
#pragma unroll
    for (int mi = 0; mi < 2; ++mi) {
#pragma unroll
        for (int g = 0; g < 8; ++g) {
            const int row = bm + m0 + mi * 16 + g + 8 * hi;
#pragma unroll
            for (int ni = 0; ni < 2; ++ni) {
                const int col = c0 + ni * 16;
                float val = acc[mi][ni][g] + (ni ? bia1 : bia0);
                if (RES) val += res[(size_t)row * N + col];
                if (ACT == 1) val = gelu_tanh_f(val);
                if (OMODE == 0) {
                    C[(size_t)row * N + col] = val;
                } else {
                    unsigned u = __float_as_uint(val);
                    unsigned hb = u >> 16;
                    unsigned lb = bf16_rne_bits(val - __uint_as_float(hb << 16));
                    size_t idx;
                    if (OMODE == 1) {
                        idx = (size_t)row * N + col;
                    } else {
                        // row = b*SEQ + s ; col = head*DK + d
                        idx = ((size_t)((row >> 11) * (NH * DK)) + col) * SEQ + (row & (SEQ - 1));
                    }
                    CH[idx] = (unsigned short)hb;
                    CL[idx] = (unsigned short)lb;
                }
            }
        }
    }
}

// ---------------------------------------------------------------------------
// bf16x3 flash attention. One block (8 waves / 256 thr) per
// (batch, head, 32-row q-tile): halves per-head K/V L2 re-staging traffic
// vs 16-row tiles. Wave w: row-half mh = w>>2, key/dim group kg = w&3.
// Q/K split-bf16 row-major; V split-bf16 pre-transposed [b][head][dim][seq].
// ---------------------------------------------------------------------------
__global__ __launch_bounds__(256) void attn_kernel(const unsigned short* __restrict__ qh,
                                                   const unsigned short* __restrict__ ql,
                                                   const unsigned short* __restrict__ kh,
                                                   const unsigned short* __restrict__ kl,
                                                   const unsigned short* __restrict__ vth,
                                                   const unsigned short* __restrict__ vtl,
                                                   float* __restrict__ ctx) {
    __shared__ __align__(16) unsigned short Kh[64][72];   // [key][dim]
    __shared__ __align__(16) unsigned short Kl[64][72];
    __shared__ __align__(16) unsigned short Vh[64][72];   // [dim][key]
    __shared__ __align__(16) unsigned short Vl[64][72];
    __shared__ __align__(16) unsigned short Ph[32][72];   // [row][key]
    __shared__ __align__(16) unsigned short Pl[32][72];
    __shared__ float redA[4][32];
    __shared__ float redB[4][32];

    const int bid = blockIdx.x;              // 0..2047
    const int qt = bid & 63;                 // SEQ/32 q-tiles
    const int head = (bid >> 6) & 15;
    const int b = bid >> 10;
    const int qbase = qt * 32;
    const size_t rowbase = (size_t)(b * SEQ + qbase);

    const int tid = threadIdx.x;
    const int lane = tid & 31, w = tid >> 5;
    const int lo = lane & 15, hi = lane >> 4;
    const int mh = w >> 2;                   // row-half (0/1)
    const int kg = w & 3;                    // key/dim group (0..3)

    // Q fragments (A layout) for this wave's 16 rows, 2 K-slabs x {hi,lo}
    Frag16 qfh[2], qfl[2];
    {
        const unsigned short* Qh = qh + (rowbase + mh * 16 + lo) * BDIM + head * DK;
        const unsigned short* Ql = ql + (rowbase + mh * 16 + lo) * BDIM + head * DK;
#pragma unroll
        for (int sl = 0; sl < 2; ++sl) {
            qfh[sl].q[0] = *(const u32x4*)(Qh + sl * 32 + 8 * hi);
            qfh[sl].q[1] = *(const u32x4*)(Qh + sl * 32 + 16 + 8 * hi);
            qfl[sl].q[0] = *(const u32x4*)(Ql + sl * 32 + 8 * hi);
            qfl[sl].q[1] = *(const u32x4*)(Ql + sl * 32 + 16 + 8 * hi);
        }
    }

    v8f cacc = {};
    float rm[8], rs[8];
#pragma unroll
    for (int g = 0; g < 8; ++g) { rm[g] = -1e30f; rs[g] = 0.0f; }
    const float scale = 0.125f; // 1/sqrt(64)

    for (int j = 0; j < SEQ / 64; ++j) {
        __syncthreads();
        {
            const int r = tid >> 2;            // key for K, dim for Vt (0..63)
            const int c0 = (tid & 3) * 16;
            const unsigned short* ksh = kh + (size_t)(b * SEQ + j * 64 + r) * BDIM + head * DK + c0;
            const unsigned short* ksl = kl + (size_t)(b * SEQ + j * 64 + r) * BDIM + head * DK + c0;
            const unsigned short* vsh = vth + ((size_t)(b * NH + head) * DK + r) * SEQ + j * 64 + c0;
            const unsigned short* vsl = vtl + ((size_t)(b * NH + head) * DK + r) * SEQ + j * 64 + c0;
#pragma unroll
            for (int i = 0; i < 2; ++i) {
                *(u32x4*)(&Kh[r][c0 + i * 8]) = *(const u32x4*)(ksh + i * 8);
                *(u32x4*)(&Kl[r][c0 + i * 8]) = *(const u32x4*)(ksl + i * 8);
                *(u32x4*)(&Vh[r][c0 + i * 8]) = *(const u32x4*)(vsh + i * 8);
                *(u32x4*)(&Vl[r][c0 + i * 8]) = *(const u32x4*)(vsl + i * 8);
            }
        }
        __syncthreads();

        // scores: this wave's 16 rows x 16 keys (keys kg*16..kg*16+15)
        v8f s = {};
#pragma unroll
        for (int sl = 0; sl < 2; ++sl) {
            Frag16 kfh, kfl;
            const unsigned short* ph = &Kh[kg * 16 + lo][sl * 32 + 8 * hi];
            const unsigned short* pl = &Kl[kg * 16 + lo][sl * 32 + 8 * hi];
            kfh.q[0] = *(const u32x4*)ph;
            kfh.q[1] = *(const u32x4*)(ph + 16);
            kfl.q[0] = *(const u32x4*)pl;
            kfl.q[1] = *(const u32x4*)(pl + 16);
            s = WMMA_BF16(qfh[sl].v, kfh.v, s);
            s = WMMA_BF16(qfl[sl].v, kfh.v, s);
            s = WMMA_BF16(qfh[sl].v, kfl.v, s);
        }

        float wmax[8];
#pragma unroll
        for (int g = 0; g < 8; ++g) {
            float x = s[g] * scale;
            s[g] = x;
#pragma unroll
            for (int mm = 8; mm > 0; mm >>= 1) x = fmaxf(x, __shfl_xor(x, mm, 16));
            wmax[g] = x;
        }
#pragma unroll
        for (int g = 0; g < 8; ++g)
            if (lo == g) redA[kg][mh * 16 + hi * 8 + g] = wmax[g];
        __syncthreads();

        float mnew[8], corr[8];
#pragma unroll
        for (int g = 0; g < 8; ++g) {
            const int r = mh * 16 + hi * 8 + g;
            float tm = fmaxf(fmaxf(redA[0][r], redA[1][r]),
                             fmaxf(redA[2][r], redA[3][r]));
            tm = fmaxf(tm, rm[g]);
            corr[g] = __expf(rm[g] - tm);
            rm[g] = tm;
            mnew[g] = tm;
        }

        float wsum[8];
#pragma unroll
        for (int g = 0; g < 8; ++g) {
            float p = __expf(s[g] - mnew[g]);
            unsigned hb = __float_as_uint(p) >> 16;
            unsigned lb = bf16_rne_bits(p - __uint_as_float(hb << 16));
            Ph[mh * 16 + hi * 8 + g][kg * 16 + lo] = (unsigned short)hb;
            Pl[mh * 16 + hi * 8 + g][kg * 16 + lo] = (unsigned short)lb;
            float x = p;
#pragma unroll
            for (int mm = 8; mm > 0; mm >>= 1) x += __shfl_xor(x, mm, 16);
            wsum[g] = x;
        }
#pragma unroll
        for (int g = 0; g < 8; ++g)
            if (lo == g) redB[kg][mh * 16 + hi * 8 + g] = wsum[g];
        __syncthreads();

#pragma unroll
        for (int g = 0; g < 8; ++g) {
            const int r = mh * 16 + hi * 8 + g;
            const float ts = redB[0][r] + redB[1][r] + redB[2][r] + redB[3][r];
            rs[g] = rs[g] * corr[g] + ts;
            cacc[g] *= corr[g];
        }

        // ctx += P @ V; this wave: rows mh*16.., dims kg*16..kg*16+15
#pragma unroll
        for (int sl = 0; sl < 2; ++sl) {
            Frag16 pfh, pfl, vfh, vfl;
            const unsigned short* pph = &Ph[mh * 16 + lo][sl * 32 + 8 * hi];
            const unsigned short* ppl = &Pl[mh * 16 + lo][sl * 32 + 8 * hi];
            pfh.q[0] = *(const u32x4*)pph;
            pfh.q[1] = *(const u32x4*)(pph + 16);
            pfl.q[0] = *(const u32x4*)ppl;
            pfl.q[1] = *(const u32x4*)(ppl + 16);
            const unsigned short* pvh = &Vh[kg * 16 + lo][sl * 32 + 8 * hi];
            const unsigned short* pvl = &Vl[kg * 16 + lo][sl * 32 + 8 * hi];
            vfh.q[0] = *(const u32x4*)pvh;
            vfh.q[1] = *(const u32x4*)(pvh + 16);
            vfl.q[0] = *(const u32x4*)pvl;
            vfl.q[1] = *(const u32x4*)(pvl + 16);
            cacc = WMMA_BF16(pfh.v, vfh.v, cacc);
            cacc = WMMA_BF16(pfl.v, vfh.v, cacc);
            cacc = WMMA_BF16(pfh.v, vfl.v, cacc);
        }
    }

    float* op = ctx + rowbase * BDIM + head * DK;
#pragma unroll
    for (int g = 0; g < 8; ++g) {
        const int r = mh * 16 + g + 8 * hi;
        op[(size_t)r * BDIM + kg * 16 + lo] = cacc[g] / rs[g];
    }
}

// ---------------------------------------------------------------------------
// Launch. Workspace (float units), SZ = 4096*1024:
//   [0,SZ)      h (f32)  -> reused as ctx
//   [SZ,4SZ)    qh,ql,kh,kl,vth,vtl (bf16, SZ ushorts each) -> reused as h2
//   [4SZ,5SZ)   x1 (f32)
//   [5SZ,9SZ)   ff1 (f32)                          total 9*SZ ~ 151 MB
// ---------------------------------------------------------------------------
extern "C" void kernel_launch(void* const* d_in, const int* in_sizes, int n_in,
                              void* d_out, int out_size, void* d_ws, size_t ws_size,
                              hipStream_t stream) {
    (void)in_sizes; (void)n_in; (void)out_size; (void)ws_size;
    const float* x   = (const float*)d_in[0];
    const float* Wq  = (const float*)d_in[1];
    const float* bq  = (const float*)d_in[2];
    const float* Wk  = (const float*)d_in[3];
    const float* bk  = (const float*)d_in[4];
    const float* Wv  = (const float*)d_in[5];
    const float* bv  = (const float*)d_in[6];
    const float* Wo  = (const float*)d_in[7];
    const float* bo  = (const float*)d_in[8];
    const float* W1  = (const float*)d_in[9];
    const float* b1  = (const float*)d_in[10];
    const float* W2  = (const float*)d_in[11];
    const float* b2  = (const float*)d_in[12];
    const float* g1  = (const float*)d_in[13];
    const float* be1 = (const float*)d_in[14];
    const float* g2  = (const float*)d_in[15];
    const float* be2 = (const float*)d_in[16];
    // d_in[17] = mask (0, unused)

    float* out = (float*)d_out;
    float* ws = (float*)d_ws;
    const size_t SZ = (size_t)MROWS * BDIM;

    float* h   = ws;
    unsigned short* u = (unsigned short*)(ws + SZ);
    unsigned short* qh_ = u;
    unsigned short* ql_ = u + 1 * SZ;
    unsigned short* kh_ = u + 2 * SZ;
    unsigned short* kl_ = u + 3 * SZ;
    unsigned short* vth_ = u + 4 * SZ;
    unsigned short* vtl_ = u + 5 * SZ;
    float* x1  = ws + 4 * SZ;
    float* ff1 = ws + 5 * SZ;  // 4*SZ floats
    float* ctxb = h;           // h dead after QKV projections
    float* h2   = ws + SZ;     // q/k splits dead after attention

    const dim3 gD(BDIM / 64, MROWS / 128);

    ln_kernel<<<MROWS, 256, 0, stream>>>(x, g1, be1, h);
    gemm_bf16x3_kernel<0, false, 1><<<gD, 256, 0, stream>>>(h, Wq, bq, nullptr, nullptr, qh_, ql_, MROWS, BDIM, BDIM);
    gemm_bf16x3_kernel<0, false, 1><<<gD, 256, 0, stream>>>(h, Wk, bk, nullptr, nullptr, kh_, kl_, MROWS, BDIM, BDIM);
    gemm_bf16x3_kernel<0, false, 2><<<gD, 256, 0, stream>>>(h, Wv, bv, nullptr, nullptr, vth_, vtl_, MROWS, BDIM, BDIM);
    attn_kernel<<<NB * NH * (SEQ / 32), 256, 0, stream>>>(qh_, ql_, kh_, kl_, vth_, vtl_, ctxb);
    gemm_bf16x3_kernel<0, true, 0><<<gD, 256, 0, stream>>>(ctxb, Wo, bo, x, x1, nullptr, nullptr, MROWS, BDIM, BDIM);
    ln_kernel<<<MROWS, 256, 0, stream>>>(x1, g2, be2, h2);
    gemm_bf16x3_kernel<1, false, 0><<<dim3(DFF / 64, MROWS / 128), 256, 0, stream>>>(h2, W1, b1, nullptr, ff1, nullptr, nullptr, MROWS, DFF, BDIM);
    gemm_bf16x3_kernel<0, true, 0><<<gD, 256, 0, stream>>>(ff1, W2, b2, x1, out, nullptr, nullptr, MROWS, BDIM, DFF);
}
